// AffinityEnergyLoss_6708738916949
// MI455X (gfx1250) — matrix operand order repaired and checked by reading, not compiled, not run on hardware
//
#include <hip/hip_runtime.h>
#include <hip/hip_bf16.h>
#include <stdint.h>

typedef __attribute__((ext_vector_type(16))) _Float16 v16h;
typedef __attribute__((ext_vector_type(8)))  _Float16 v8h;
typedef __attribute__((ext_vector_type(8)))  float    v8f;

#define BZ    4
#define NC    21
#define TOKN  1024
#define HEADS 8
#define TOTN  1025
#define PADK  1032   // 1024 + 8 halfs of padding -> 2064B row pitch (16B aligned, bank-spread)

// ---------------------------------------------------------------------------
// Kernel 1: channel softmax of preds -> prob (f32, [b][t][c]) and transposed,
// zero-padded f16 probT ([b][c:32][k]) for the WMMA B operand. Zero loss accum.
// ---------------------------------------------------------------------------
__global__ void k_softmax_pack(const float* __restrict__ preds,
                               float* __restrict__ probF32,
                               _Float16* __restrict__ probHT,
                               float* __restrict__ lossAcc) {
    int idx = blockIdx.x * blockDim.x + threadIdx.x;
    if (idx == 0) { lossAcc[0] = 0.f; lossAcc[1] = 0.f; }
    if (idx >= BZ * TOKN) return;
    int b = idx >> 10, t = idx & 1023;
    const float* p = preds + (size_t)b * NC * TOKN + t;
    float m = -3.4e38f;
    #pragma unroll
    for (int c = 0; c < NC; ++c) m = fmaxf(m, p[(size_t)c * TOKN]);
    float e[NC]; float s = 0.f;
    #pragma unroll
    for (int c = 0; c < NC; ++c) { e[c] = __expf(p[(size_t)c * TOKN] - m); s += e[c]; }
    float inv = 1.0f / s;
    float* pf = probF32 + (size_t)idx * NC;
    _Float16* ph = probHT + (size_t)b * 32 * TOKN + t;
    #pragma unroll
    for (int c = 0; c < NC; ++c) {
        float v = e[c] * inv;
        pf[c] = v;
        ph[(size_t)c * TOKN] = (_Float16)v;
    }
    #pragma unroll
    for (int c = NC; c < 32; ++c) ph[(size_t)c * TOKN] = (_Float16)0.f;  // pad B cols
}

// ---------------------------------------------------------------------------
// Kernel 2: streaming fused attention-average x prob matmul + loss.
// grid = (64 row tiles, 4 batches), block = 1024 (32 waves of 32).
// Each wave handles exactly one of the 32 (layer,head,enc/dec) matrices
// for a 16-row tile; K=1024 streamed in chunks of 32 through WMMA.
// ---------------------------------------------------------------------------
__global__ void __launch_bounds__(1024, 1)
k_affinity_wmma(const float* __restrict__ attns,     // (2,4,8,1025,1025)
                const float* __restrict__ dattns,    // (2,4,8,1024,1024)
                const _Float16* __restrict__ probHT, // (4,32,1024) f16
                const float* __restrict__ probF32,   // (4,1024,21) f32
                const int* __restrict__ rois,        // (4,1024)
                float* __restrict__ lossAcc) {
    __shared__ _Float16 probLDS[32 * PADK];   // B operand, transposed [col][k]
    __shared__ float    accLDS[16 * 33];      // 16 rows x 32 cols (pitch 33)

    const int tile = blockIdx.x;        // 0..63  (16 token rows each)
    const int b    = blockIdx.y;        // 0..3
    const int row0 = tile * 16;
    const int tid  = threadIdx.x;
    const int lane = tid & 31;
    const int wv   = tid >> 5;          // 0..31 -> matrix index

    // Stage B into LDS once per workgroup (64 KB, u32 copies).
    {
        const uint32_t* src = (const uint32_t*)__builtin_assume_aligned(
            probHT + (size_t)b * 32 * TOKN, 16);
        for (int i = tid; i < 32 * (TOKN / 2); i += 1024) {
            int c = i / (TOKN / 2), kw = i - c * (TOKN / 2);
            *(uint32_t*)&probLDS[c * PADK + kw * 2] = src[(size_t)c * (TOKN / 2) + kw];
        }
    }
    if (tid < 16 * 33) accLDS[tid] = 0.f;
    __syncthreads();

    // A layout (16-bit A 16x32): lane<16 -> row=lane, K = 0..7 & 16..23;
    // lane>=16 -> row=lane-16, K = 8..15 & 24..31.
    const int rowL  = lane & 15;
    const int khalf = (lane >> 4) << 3;   // 0 or 8
    const int koffB = (lane >> 4) << 3;   // 0 or 8 (B: K = koffB..+7 & 16+koffB..+7, same as A)
    const int col   = lane & 15;

    float rowpart = 0.f;
    v8f cm0 = {};   // cols 0..15
    v8f cm1 = {};   // cols 16..31

    // One K-chunk step: load 16x32 A tile (f32->f16 + row-sum), B from LDS, 2 WMMAs.
    auto step = [&](const float* rowPtr, int kc) {
        v16h a;
        const float* ap = rowPtr + kc + khalf;
        __builtin_prefetch(ap + 64, 0, 0);           // ~2 chunks ahead, streaming
        #pragma unroll
        for (int i = 0; i < 8; ++i) {
            float x = ap[i];
            float y = ap[16 + i];
            a[i]     = (_Float16)x;
            a[8 + i] = (_Float16)y;
            rowpart += x + y;
        }
        const int kb = kc + koffB;
        const v8h* bp0 = (const v8h*)&probLDS[col * PADK + kb];
        const v8h* bp1 = (const v8h*)&probLDS[(col + 16) * PADK + kb];
        v8h p0 = bp0[0], p1 = bp0[2];
        v8h q0 = bp1[0], q1 = bp1[2];
        v16h b0 = __builtin_shufflevector(p0, p1, 0,1,2,3,4,5,6,7,8,9,10,11,12,13,14,15);
        v16h b1 = __builtin_shufflevector(q0, q1, 0,1,2,3,4,5,6,7,8,9,10,11,12,13,14,15);
        cm0 = __builtin_amdgcn_wmma_f32_16x16x32_f16(false, a, false, b0,
                                                     (short)0, cm0, false, false);
        cm1 = __builtin_amdgcn_wmma_f32_16x16x32_f16(false, a, false, b1,
                                                     (short)0, cm1, false, false);
    };

    // Wave-uniform encoder/decoder split (separate loops so the decoder side
    // gets provable 16B alignment -> global_load_b128; encoder stride is 1025
    // and inherently 4B-aligned).
    if (wv < 16) {
        const int l = wv >> 3, hd = wv & 7;
        size_t mat = ((size_t)l * BZ + b) * HEADS + hd;
        const float* rowPtr = attns + mat * (size_t)TOTN * TOTN
                            + (size_t)(row0 + 1 + rowL) * TOTN + 1;  // [1:,1:]
        for (int kc = 0; kc < TOKN; kc += 32) step(rowPtr, kc);
    } else {
        const int mm = wv - 16;
        const int l = mm >> 3, hd = mm & 7;
        size_t mat = ((size_t)l * BZ + b) * HEADS + hd;
        const float* base = (const float*)__builtin_assume_aligned(dattns, 16);
        const float* rowPtr = base + mat * (size_t)TOKN * TOKN
                            + (size_t)(row0 + rowL) * TOKN;
        for (int kc = 0; kc < TOKN; kc += 32) step(rowPtr, kc);
    }

    // Full row sum: lanes L and L^16 hold the two K-halves of row L&15.
    float rs = rowpart + __shfl_xor(rowpart, 16, 32);

    // Scale this matrix's contribution by 1/(32*rowsum[row]) and reduce across
    // all 32 waves into LDS (ds_add_f32).
    // C layout: VGPR r -> M = r (lanes 0-15) or r+8 (lanes 16-31), N = lane&15.
    #pragma unroll
    for (int r = 0; r < 8; ++r) {
        int M = r + ((lane >> 4) << 3);
        float rsM = __shfl(rs, M, 32);       // lane M (<16) holds rowsum of row M
        float sc = 1.0f / (32.0f * rsM);
        int N = lane & 15;
        atomicAdd(&accLDS[M * 33 + N],      cm0[r] * sc);
        atomicAdd(&accLDS[M * 33 + 16 + N], cm1[r] * sc);
    }
    __syncthreads();

    // Finalize 16 rows: normalize affinity row, masked L1 vs prob, global atomics.
    if (tid < 16) {
        const int t = row0 + tid;
        float aff[NC]; float asum = 0.f;
        #pragma unroll
        for (int c = 0; c < NC; ++c) { aff[c] = accLDS[tid * 33 + c]; asum += aff[c]; }
        float ainv = 1.0f / asum;
        const float* pf = probF32 + ((size_t)b * TOKN + t) * NC;
        float diff = 0.f;
        #pragma unroll
        for (int c = 0; c < NC; ++c) diff += fabsf(pf[c] - aff[c] * ainv);
        if (rois[b * TOKN + t] != 0) {
            atomicAdd(&lossAcc[0], diff);
            atomicAdd(&lossAcc[1], 1.0f);
        }
    }
}

// ---------------------------------------------------------------------------
// Kernel 3: final scalar.
// ---------------------------------------------------------------------------
__global__ void k_finalize(const float* __restrict__ lossAcc, float* __restrict__ out) {
    float L = lossAcc[0], Nf = lossAcc[1];
    out[0] = (Nf > 0.f) ? (L / Nf) : L;   // WEIGHT = 1.0
}

extern "C" void kernel_launch(void* const* d_in, const int* in_sizes, int n_in,
                              void* d_out, int out_size, void* d_ws, size_t ws_size,
                              hipStream_t stream) {
    const float* preds  = (const float*)d_in[0];
    const int*   rois   = (const int*)d_in[3];
    const float* attns  = (const float*)d_in[5];
    const float* dattns = (const float*)d_in[6];

    char* ws = (char*)d_ws;
    float*    probF32 = (float*)ws;                           // 4*1024*21*4 = 344064 B
    _Float16* probHT  = (_Float16*)(ws + 344064);             // 4*32*1024*2 = 262144 B
    float*    lossAcc = (float*)(ws + 344064 + 262144);       // 2 floats

    k_softmax_pack<<<dim3((BZ * TOKN + 255) / 256), dim3(256), 0, stream>>>(
        preds, probF32, probHT, lossAcc);

    k_affinity_wmma<<<dim3(TOKN / 16, BZ), dim3(1024), 0, stream>>>(
        attns, dattns, probHT, probF32, rois, lossAcc);

    k_finalize<<<1, 1, 0, stream>>>(lossAcc, (float*)d_out);
}
